// EncoderSingleLayer_76897094468068
// MI455X (gfx1250) — matrix operand order repaired and checked by reading, not compiled, not run on hardware
//
#include <hip/hip_runtime.h>
#include <hip/hip_bf16.h>
#include <stdint.h>

typedef __attribute__((ext_vector_type(16))) __bf16 v16bf;
typedef __attribute__((ext_vector_type(8)))  float  v8f;
typedef unsigned int u32x4 __attribute__((ext_vector_type(4)));
typedef int          i32x8 __attribute__((ext_vector_type(8)));
typedef int          i32x4 __attribute__((ext_vector_type(4)));

#define DEV static __device__ __forceinline__

// ---------- helpers ----------
DEV uint16_t f2bf(float f) {
  union { float f; uint32_t u; } x; x.f = f;
  uint32_t r = x.u + 0x7FFFu + ((x.u >> 16) & 1u);   // round-to-nearest-even
  return (uint16_t)(r >> 16);
}

DEV v8f vzero() {
  v8f z;
#pragma unroll
  for (int i = 0; i < 8; ++i) z[i] = 0.f;
  return z;
}

// A-fragment (16x32, MxK) per ISA 7.12.2: lanes 0-15 row M=lane, K {0..7,16..23};
// lanes 16-31 same rows, K {8..15,24..31}. Source: row-major [row][k] bf16 LDS tile.
DEV v16bf load_fragA(const uint16_t* t, int ld, int row0, int k0) {
  int lane = threadIdx.x & 31;
  int r = lane & 15, hi = lane >> 4;
  union { uint32_t u[8]; v16bf v; } f;
  const uint16_t* base = t + (size_t)(row0 + r) * ld + k0 + hi * 8;
#pragma unroll
  for (int i = 0; i < 8; ++i) {
    int k = (i & 3) * 2 + (i >> 2) * 16;
    f.u[i] = *reinterpret_cast<const uint32_t*>(base + k);
  }
  return f.v;
}

// B-fragment (32x16, KxN): lanes 0-15 col N=lane, K {0..7,16..23};
// lanes 16-31 same cols, K {8..15,24..31} (same k order as A-fragment).
// Source: transposed (N x K) row-major bf16 LDS tile.
DEV v16bf load_fragB(const uint16_t* t, int ld, int n0, int k0) {
  int lane = threadIdx.x & 31;
  int c = lane & 15, hi = lane >> 4;
  union { uint32_t u[8]; v16bf v; } f;
  const uint16_t* base = t + (size_t)(n0 + c) * ld + k0 + hi * 8;
#pragma unroll
  for (int i = 0; i < 8; ++i)
    f.u[i] = *reinterpret_cast<const uint32_t*>(base + (i & 3) * 2 + (i >> 2) * 16);
  return f.v;
}

DEV v8f wmma_bf16(v16bf a, v16bf b, v8f c) {
  return __builtin_amdgcn_wmma_f32_16x16x32_bf16(false, a, false, b, (short)0, c,
                                                 false, false);
}

// ---------- Tensor Data Mover: 2D bf16 tile (global -> LDS), per ISA ch.8 D# ----------
// tile_d0 = contiguous elements per row, tile_d1 = rows, stride0 = row pitch (elems).
DEV void tdm_load_tile(void* lds_ptr, const void* gptr, uint32_t tile_d0,
                       uint32_t tile_d1, uint64_t stride0) {
  uint64_t ga = (uint64_t)(uintptr_t)gptr;
  uint32_t lds = (uint32_t)(uintptr_t)lds_ptr;      // LDS aperture: low 32 = offset
  const uint32_t td0 = 1u << 30, td1 = 1u << 30;    // tensor dims: never clip
  u32x4 g0;
  g0[0] = 1u;                                        // count=1 (valid user D#)
  g0[1] = lds;                                       // lds_addr (bytes)
  g0[2] = (uint32_t)ga;                              // global_addr[31:0]
  g0[3] = ((uint32_t)(ga >> 32) & 0x01FFFFFFu) | (2u << 30);  // addr[56:32]|type=2
  i32x8 g1;
  g1[0] = (int)(1u << 16);                           // data_size=1 -> 2-byte elems
  g1[1] = (int)((td0 & 0xFFFFu) << 16);              // tensor_dim0 lo16
  g1[2] = (int)((td0 >> 16) | ((td1 & 0xFFFFu) << 16));   // d0 hi16 | d1 lo16
  g1[3] = (int)((td1 >> 16) | (tile_d0 << 16));      // d1 hi16 | tile_dim0
  g1[4] = (int)(tile_d1 & 0xFFFFu);                  // tile_dim1 ; tile_dim2=0
  g1[5] = (int)(uint32_t)(stride0 & 0xFFFFFFFFu);    // tensor_dim0_stride lo32
  g1[6] = (int)(uint32_t)((stride0 >> 32) & 0xFFFFu);// stride0 hi16 ; stride1=0
  g1[7] = 0;
  i32x4 z4; z4[0] = z4[1] = z4[2] = z4[3] = 0;       // groups 2/3 unused (2D)
  i32x8 z8;
#pragma unroll
  for (int i = 0; i < 8; ++i) z8[i] = 0;
  __builtin_amdgcn_tensor_load_to_lds(g0, g1, z4, z4, z8, 0);
}

// ---------- f32 -> bf16 convert ----------
__global__ void cvt_kernel(const float* __restrict__ in, uint16_t* __restrict__ out,
                           int n2) {
  int i = blockIdx.x * blockDim.x + threadIdx.x;   // each thread packs 2 elems
  if (i < n2) {
    float a = in[2 * i + 0], b = in[2 * i + 1];
    uint32_t u = (uint32_t)f2bf(a) | ((uint32_t)f2bf(b) << 16);
    *reinterpret_cast<uint32_t*>(out + 2 * i) = u;
  }
}

// ---------- one-time weight transpose+convert: f32 (KxN or (H,K,64)) -> bf16 NxK ----------
template <bool QKV>
__global__ void wtr_kernel(const float* __restrict__ in, uint16_t* __restrict__ out,
                           int N, int K) {
  int idx = blockIdx.x * blockDim.x + threadIdx.x;
  if (idx >= N * K) return;
  int n = idx / K, k = idx - n * K;
  size_t src = QKV ? ((size_t)(n >> 6) * K * 64 + (size_t)k * 64 + (n & 63))
                   : ((size_t)k * N + n);
  out[idx] = f2bf(in[src]);
}

// ---------- WMMA GEMM, TDM-fed, double-buffered: C = A(MxK) * Bt(NxK)^T ----------
enum { MODE_QKV = 0, MODE_RESID = 1, MODE_RELU = 2 };

template <int MODE>
__global__ __launch_bounds__(256) void gemm_kernel(
    const uint16_t* __restrict__ A, const uint16_t* __restrict__ Bt,
    const float* __restrict__ bias, const float* __restrict__ resid,
    void* __restrict__ out, int M, int N, int K) {
  __shared__ alignas(16) uint16_t As[2][128 * 32];   // [m][k]
  __shared__ alignas(16) uint16_t Bs[2][128 * 32];   // [n][k]
  const int tid = threadIdx.x;
  const int wave = tid >> 5, wm = wave >> 2, wn = wave & 3;
  const int n0 = blockIdx.x * 128, m0 = blockIdx.y * 128;
  const int nk = K >> 5;

  v8f acc[4][2];
#pragma unroll
  for (int i = 0; i < 4; ++i)
#pragma unroll
    for (int j = 0; j < 2; ++j) acc[i][j] = vzero();

  // prologue: DMA tile 0
  if (wave == 0) {
    tdm_load_tile(&As[0][0], A + (size_t)m0 * K, 32, 128, (uint64_t)K);
    tdm_load_tile(&Bs[0][0], Bt + (size_t)n0 * K, 32, 128, (uint64_t)K);
    __builtin_amdgcn_s_wait_tensorcnt(0);
  }
  __syncthreads();

  for (int j = 0; j < nk; ++j) {
    const int cur = j & 1;
    if (wave == 0 && j + 1 < nk) {                   // prefetch next tiles via TDM
      const int kk = (j + 1) << 5;
      tdm_load_tile(&As[cur ^ 1][0], A + (size_t)m0 * K + kk, 32, 128, (uint64_t)K);
      tdm_load_tile(&Bs[cur ^ 1][0], Bt + (size_t)n0 * K + kk, 32, 128, (uint64_t)K);
    }
    v16bf af[4], bfr[2];
#pragma unroll
    for (int mt = 0; mt < 4; ++mt)
      af[mt] = load_fragA(&As[cur][0], 32, wm * 64 + mt * 16, 0);
#pragma unroll
    for (int nt = 0; nt < 2; ++nt)
      bfr[nt] = load_fragB(&Bs[cur][0], 32, wn * 32 + nt * 16, 0);
#pragma unroll
    for (int mt = 0; mt < 4; ++mt)
#pragma unroll
      for (int nt = 0; nt < 2; ++nt)
        acc[mt][nt] = wmma_bf16(af[mt], bfr[nt], acc[mt][nt]);

    if (wave == 0) __builtin_amdgcn_s_wait_tensorcnt(0);
    __syncthreads();
  }

  const int lane = tid & 31, ln = lane & 15, hi = lane >> 4;
#pragma unroll
  for (int mt = 0; mt < 4; ++mt)
#pragma unroll
    for (int nt = 0; nt < 2; ++nt)
#pragma unroll
      for (int r = 0; r < 8; ++r) {
        int m = m0 + wm * 64 + mt * 16 + r + hi * 8;
        int n = n0 + wn * 32 + nt * 16 + ln;
        float v = acc[mt][nt][r] + bias[n];
        if (MODE == MODE_QKV) {
          int b = m >> 11, s = m & 2047, h = n >> 6, dk = n & 63;
          ((uint16_t*)out)[((size_t)(b * 16 + h) * 2048 + s) * 64 + dk] = f2bf(v);
        } else if (MODE == MODE_RELU) {
          ((uint16_t*)out)[(size_t)m * N + n] = f2bf(v > 0.f ? v : 0.f);
        } else {                         // bias + residual -> f32
          ((float*)out)[(size_t)m * N + n] = v + resid[(size_t)m * N + n];
        }
      }
}

// ---------- flash attention: 1 block = 4 waves = 64 queries of one (b,h) ----------
__global__ __launch_bounds__(128) void attn_kernel(
    const uint16_t* __restrict__ Q, const uint16_t* __restrict__ Kb,
    const uint16_t* __restrict__ V, uint16_t* __restrict__ ctx) {
  __shared__ alignas(16) uint16_t Qs[64 * 64];   // [q][dk]
  __shared__ alignas(16) uint16_t Ks[64 * 64];   // [key][dk]  (= N x K for B-frag)
  __shared__ alignas(16) uint16_t Vt[64 * 64];   // [dv][key]  (= N x K for B-frag)
  __shared__ alignas(16) uint16_t Ps[64 * 64];   // [q][key]
  const int tid = threadIdx.x;
  const int w = tid >> 5, lane = tid & 31, ln = lane & 15, hi = lane >> 4;
  const int bh = blockIdx.y;                     // b*16 + h
  const int q0 = blockIdx.x * 64;
  const size_t base = (size_t)bh * 2048 * 64;

  // stage Q tile once
#pragma unroll
  for (int i = 0; i < 4; ++i) {
    int idx = i * 128 + tid;
    int row = idx >> 3, qq = idx & 7;
    *reinterpret_cast<uint4*>(&Qs[row * 64 + qq * 8]) =
        *reinterpret_cast<const uint4*>(Q + base + (size_t)(q0 + row) * 64 + qq * 8);
  }

  float run_max[8], run_sum[8];
#pragma unroll
  for (int r = 0; r < 8; ++r) { run_max[r] = -1e30f; run_sum[r] = 0.f; }
  v8f of[4];
#pragma unroll
  for (int nt = 0; nt < 4; ++nt) of[nt] = vzero();

  for (int j = 0; j < 32; ++j) {
    __syncthreads();                             // prev iter done with Ks/Vt
    const int k0 = j * 64;
    if (w == 0)                                  // K tile via TDM (64x64 bf16)
      tdm_load_tile(&Ks[0], Kb + base + (size_t)k0 * 64, 64, 64, 64);
#pragma unroll
    for (int i = 0; i < 16; ++i) {               // V transposed -> Vt[dv][key]
      int e = i * 128 + tid;
      int row = e >> 5, pr = e & 31;
      uint32_t u = *reinterpret_cast<const uint32_t*>(
          V + base + (size_t)(k0 + row) * 64 + pr * 2);
      Vt[(pr * 2 + 0) * 64 + row] = (uint16_t)(u & 0xFFFFu);
      Vt[(pr * 2 + 1) * 64 + row] = (uint16_t)(u >> 16);
    }
    if (w == 0) __builtin_amdgcn_s_wait_tensorcnt(0);
    __syncthreads();

    // S = Q K^T for rows [w*16, w*16+16), all 64 key cols
    v16bf a0 = load_fragA(Qs, 64, w * 16, 0);
    v16bf a1 = load_fragA(Qs, 64, w * 16, 32);
    v8f sf[4];
#pragma unroll
    for (int nt = 0; nt < 4; ++nt) {
      v16bf b0 = load_fragB(Ks, 64, nt * 16, 0);
      v16bf b1 = load_fragB(Ks, 64, nt * 16, 32);
      v8f c = vzero();
      c = wmma_bf16(a0, b0, c);
      c = wmma_bf16(a1, b1, c);
      sf[nt] = c;
    }

    // online softmax (rows live in 16-lane groups)
#pragma unroll
    for (int r = 0; r < 8; ++r) {
      float mx = -1e30f;
#pragma unroll
      for (int nt = 0; nt < 4; ++nt) {
        float s = sf[nt][r] * 0.125f;            // 1/sqrt(64)
        sf[nt][r] = s;
        mx = fmaxf(mx, s);
      }
#pragma unroll
      for (int msk = 1; msk < 16; msk <<= 1) mx = fmaxf(mx, __shfl_xor(mx, msk, 32));
      float nm = fmaxf(run_max[r], mx);
      float alpha = __expf(run_max[r] - nm);
      run_max[r] = nm;
      int rowl = w * 16 + r + hi * 8;
      float psum = 0.f;
#pragma unroll
      for (int nt = 0; nt < 4; ++nt) {
        float p = __expf(sf[nt][r] - nm);
        psum += p;
        Ps[rowl * 64 + nt * 16 + ln] = f2bf(p);
      }
#pragma unroll
      for (int msk = 1; msk < 16; msk <<= 1) psum += __shfl_xor(psum, msk, 32);
      run_sum[r] = run_sum[r] * alpha + psum;
#pragma unroll
      for (int nt = 0; nt < 4; ++nt) of[nt][r] *= alpha;
    }

    // O += P * V  (wave reads only its own Ps rows; LDS in-order per wave)
#pragma unroll
    for (int ks = 0; ks < 2; ++ks) {
      v16bf pa = load_fragA(Ps, 64, w * 16, ks * 32);
#pragma unroll
      for (int nt = 0; nt < 4; ++nt) {
        v16bf vb = load_fragB(Vt, 64, nt * 16, ks * 32);
        of[nt] = wmma_bf16(pa, vb, of[nt]);
      }
    }
  }

  // write ctx as (b, s, h*64+dv) bf16
  const int b = bh >> 4, h = bh & 15;
#pragma unroll
  for (int r = 0; r < 8; ++r) {
    float inv = 1.f / run_sum[r];
    int s = q0 + w * 16 + r + hi * 8;
#pragma unroll
    for (int nt = 0; nt < 4; ++nt) {
      int dv = nt * 16 + ln;
      ctx[((size_t)b * 2048 + s) * 1024 + h * 64 + dv] = f2bf(of[nt][r] * inv);
    }
  }
}

// ---------- LayerNorm over rows of 1024 ----------
__global__ __launch_bounds__(256) void ln_kernel(
    const float* __restrict__ in, const float* __restrict__ g,
    const float* __restrict__ be, float* __restrict__ outf,
    uint16_t* __restrict__ outb) {
  __shared__ float red[256];
  const int row = blockIdx.x, tid = threadIdx.x;
  const float* x = in + (size_t)row * 1024;
  float v[4], s = 0.f;
#pragma unroll
  for (int i = 0; i < 4; ++i) { v[i] = x[tid + i * 256]; s += v[i]; }
  red[tid] = s; __syncthreads();
  for (int o = 128; o > 0; o >>= 1) { if (tid < o) red[tid] += red[tid + o]; __syncthreads(); }
  float mean = red[0] * (1.f / 1024.f);
  __syncthreads();
  s = 0.f;
#pragma unroll
  for (int i = 0; i < 4; ++i) { float d = v[i] - mean; s += d * d; }
  red[tid] = s; __syncthreads();
  for (int o = 128; o > 0; o >>= 1) { if (tid < o) red[tid] += red[tid + o]; __syncthreads(); }
  float rstd = rsqrtf(red[0] * (1.f / 1024.f) + 1e-5f);
#pragma unroll
  for (int i = 0; i < 4; ++i) {
    int c = tid + i * 256;
    float y = g[c] * (v[i] - mean) * rstd + be[c];
    outf[(size_t)row * 1024 + c] = y;
    if (outb) outb[(size_t)row * 1024 + c] = f2bf(y);
  }
}

// ---------- launcher ----------
extern "C" void kernel_launch(void* const* d_in, const int* in_sizes, int n_in,
                              void* d_out, int out_size, void* d_ws, size_t ws_size,
                              hipStream_t stream) {
  (void)in_sizes; (void)n_in; (void)out_size; (void)ws_size;
  const float* x  = (const float*)d_in[0];
  const float* Wq = (const float*)d_in[1];
  const float* bq = (const float*)d_in[2];
  const float* Wk = (const float*)d_in[3];
  const float* bk = (const float*)d_in[4];
  const float* Wv = (const float*)d_in[5];
  const float* bv = (const float*)d_in[6];
  const float* Wo = (const float*)d_in[7];
  const float* bo = (const float*)d_in[8];
  const float* W1 = (const float*)d_in[9];
  const float* b1 = (const float*)d_in[10];
  const float* W2 = (const float*)d_in[11];
  const float* b2 = (const float*)d_in[12];
  const float* g1 = (const float*)d_in[13];
  const float* be1 = (const float*)d_in[14];
  const float* g2 = (const float*)d_in[15];
  const float* be2 = (const float*)d_in[16];

  char* ws = (char*)d_ws;
  const size_t MB = 1u << 20;
  uint16_t* xb   = (uint16_t*)(ws + 0 * MB);    // 8 MB
  uint16_t* qb   = (uint16_t*)(ws + 8 * MB);    // 8 MB
  uint16_t* kb   = (uint16_t*)(ws + 16 * MB);   // 8 MB
  uint16_t* vb   = (uint16_t*)(ws + 24 * MB);   // 8 MB
  uint16_t* ctxb = (uint16_t*)(ws + 32 * MB);   // 8 MB
  float*    sum1 = (float*)   (ws + 40 * MB);   // 16 MB
  float*    y1   = (float*)   (ws + 56 * MB);   // 16 MB
  uint16_t* y1b  = (uint16_t*)(ws + 72 * MB);   // 8 MB
  uint16_t* Wqt  = (uint16_t*)(ws + 80 * MB);   // 2 MB (bf16, N x K)
  uint16_t* Wkt  = (uint16_t*)(ws + 82 * MB);   // 2 MB
  uint16_t* Wvt  = (uint16_t*)(ws + 84 * MB);   // 2 MB
  uint16_t* Wot  = (uint16_t*)(ws + 86 * MB);   // 2 MB
  uint16_t* W1t  = (uint16_t*)(ws + 88 * MB);   // 8 MB
  uint16_t* W2t  = (uint16_t*)(ws + 96 * MB);   // 8 MB
  uint16_t* ffh  = (uint16_t*)(ws + 8 * MB);    // 32 MB, reuses q/k/v/ctx
  float*    sum2 = (float*)   (ws + 40 * MB);   // reuses sum1

  dim3 b256(256), b128(128);
  const int M = 4096;                            // B*S

  cvt_kernel<<<dim3(M * 1024 / 2 / 256), b256, 0, stream>>>(x, xb, M * 1024 / 2);
  wtr_kernel<true ><<<dim3(4096),  b256, 0, stream>>>(Wq, Wqt, 1024, 1024);
  wtr_kernel<true ><<<dim3(4096),  b256, 0, stream>>>(Wk, Wkt, 1024, 1024);
  wtr_kernel<true ><<<dim3(4096),  b256, 0, stream>>>(Wv, Wvt, 1024, 1024);
  wtr_kernel<false><<<dim3(4096),  b256, 0, stream>>>(Wo, Wot, 1024, 1024);
  wtr_kernel<false><<<dim3(16384), b256, 0, stream>>>(W1, W1t, 4096, 1024);
  wtr_kernel<false><<<dim3(16384), b256, 0, stream>>>(W2, W2t, 1024, 4096);

  gemm_kernel<MODE_QKV><<<dim3(8, 32), b256, 0, stream>>>(xb, Wqt, bq, nullptr, qb, M, 1024, 1024);
  gemm_kernel<MODE_QKV><<<dim3(8, 32), b256, 0, stream>>>(xb, Wkt, bk, nullptr, kb, M, 1024, 1024);
  gemm_kernel<MODE_QKV><<<dim3(8, 32), b256, 0, stream>>>(xb, Wvt, bv, nullptr, vb, M, 1024, 1024);

  attn_kernel<<<dim3(32, 32), b128, 0, stream>>>(qb, kb, vb, ctxb);

  gemm_kernel<MODE_RESID><<<dim3(8, 32), b256, 0, stream>>>(ctxb, Wot, bo, x, sum1, M, 1024, 1024);
  ln_kernel<<<dim3(4096), b256, 0, stream>>>(sum1, g1, be1, y1, y1b);

  gemm_kernel<MODE_RELU><<<dim3(32, 32), b256, 0, stream>>>(y1b, W1t, b1, nullptr, ffh, M, 4096, 1024);
  gemm_kernel<MODE_RESID><<<dim3(8, 32), b256, 0, stream>>>(ffh, W2t, b2, y1, sum2, M, 1024, 4096);
  ln_kernel<<<dim3(4096), b256, 0, stream>>>(sum2, g2, be2, (float*)d_out, nullptr);
}
